// ED_DeepGCN_89601607729771
// MI455X (gfx1250) — matrix operand order repaired and checked by reading, not compiled, not run on hardware
//
#include <hip/hip_runtime.h>
#include <hip/hip_bf16.h>

// ---------------------------------------------------------------------------
// ED_DeepGCN for MI455X (gfx1250).
//   B=4, N=4096, K=10, EMB=1024, ED=4, CIN=1028 (padded to 1056 = 33*32)
// Heavy op: per-batch GEMM y = conv_w[1024,1028] @ z[1028,4096]  (~34.5 GFLOP)
//   -> bf16 inputs + v_wmma_f32_16x16x32_bf16 (f32 accumulate), memory-bound.
// GEMM config: block tile 128(M)x128(N), 8 waves, wave tile 32(M)x64(N)
//   = 8 WMMAs per K-step per wave against 12 ds_load_b128 (B frags reused).
// ---------------------------------------------------------------------------

#define BB   4
#define NN   4096
#define KNB  10
#define EMB  1024
#define ED   4
#define CIN  1028
#define CPAD 1056
#define BN_EPS 1e-5f

typedef __attribute__((ext_vector_type(16))) __bf16 v16bf;
typedef __attribute__((ext_vector_type(8)))  float  v8f;

__device__ __forceinline__ unsigned short f2bf(float x) {
    unsigned u = __float_as_uint(x);
    unsigned r = u + 0x7FFFu + ((u >> 16) & 1u);   // round-to-nearest-even
    return (unsigned short)(r >> 16);
}

// --------------------------- 3x3 symmetric eigvals (ascending) -------------
__device__ __forceinline__ void eig3_sym(float a00, float a01, float a02,
                                         float a11, float a12, float a22,
                                         float e[3]) {
    float p1 = a01 * a01 + a02 * a02 + a12 * a12;
    float q  = (a00 + a11 + a22) * (1.0f / 3.0f);
    float d0 = a00 - q, d1 = a11 - q, d2 = a22 - q;
    float p2 = d0 * d0 + d1 * d1 + d2 * d2 + 2.0f * p1;
    float p  = sqrtf(p2 * (1.0f / 6.0f));
    if (p < 1e-12f) { e[0] = e[1] = e[2] = q; return; }
    float inv = 1.0f / p;
    float b00 = d0 * inv, b01 = a01 * inv, b02 = a02 * inv;
    float b11 = d1 * inv, b12 = a12 * inv, b22 = d2 * inv;
    float detB = b00 * (b11 * b22 - b12 * b12)
               - b01 * (b01 * b22 - b12 * b02)
               + b02 * (b01 * b12 - b11 * b02);
    float r = fminf(1.0f, fmaxf(-1.0f, 0.5f * detB));
    float phi = acosf(r) * (1.0f / 3.0f);
    float e1 = q + 2.0f * p * cosf(phi);
    float e3 = q + 2.0f * p * cosf(phi + 2.0943951023931953f);
    float e2 = 3.0f * q - e1 - e3;
    e[0] = e3; e[1] = e2; e[2] = e1;   // ascending, matches eigvalsh
}

// ------------------- Kernel 1: KNN + cov + eig + MLP (fused) ---------------
// grid (N/256, B), block 256.  Whole point cloud lives in LDS (48 KB).
__global__ void __launch_bounds__(256)
knn_eig_mlp(const float* __restrict__ data,
            const float* __restrict__ W1, const float* __restrict__ b1,
            const float* __restrict__ W2, const float* __restrict__ b2,
            float* __restrict__ h_out,      // [B*N*4]
            float* __restrict__ data_out)   // [B*N*3] passthrough
{
    __shared__ float px[NN], py[NN], pz[NN];
    const int b = blockIdx.y;
    const float* pd = data + (size_t)b * NN * 3;
    for (int i = threadIdx.x; i < NN; i += blockDim.x) {
        px[i] = pd[i * 3 + 0];
        py[i] = pd[i * 3 + 1];
        pz[i] = pd[i * 3 + 2];
    }
    __syncthreads();

    const int q = blockIdx.x * blockDim.x + threadIdx.x;   // 0..N-1
    const float qx = px[q], qy = py[q], qz = pz[q];

    // passthrough output
    data_out[((size_t)b * NN + q) * 3 + 0] = qx;
    data_out[((size_t)b * NN + q) * 3 + 1] = qy;
    data_out[((size_t)b * NN + q) * 3 + 2] = qz;

    float bd[KNB];
    int   bi[KNB];
#pragma unroll
    for (int t = 0; t < KNB; ++t) { bd[t] = 3.0e38f; bi[t] = 0; }

    for (int j = 0; j < NN; ++j) {
        float dx = px[j] - qx, dy = py[j] - qy, dz = pz[j] - qz;
        float d = dx * dx + dy * dy + dz * dz;
        if (d < bd[KNB - 1]) {
            int p = 0;
#pragma unroll
            for (int t = 0; t < KNB - 1; ++t) if (bd[t] <= d) p = t + 1;
#pragma unroll
            for (int t = KNB - 1; t >= 0; --t) {
                if (t > p)       { bd[t] = bd[t - 1]; bi[t] = bi[t - 1]; }
                else if (t == p) { bd[t] = d;         bi[t] = j;         }
            }
        }
    }

    // covariance of the 10 nearest neighbours
    float mx = 0.f, my = 0.f, mz = 0.f;
#pragma unroll
    for (int t = 0; t < KNB; ++t) { mx += px[bi[t]]; my += py[bi[t]]; mz += pz[bi[t]]; }
    const float ik = 1.0f / (float)KNB;
    mx *= ik; my *= ik; mz *= ik;

    float c00 = 0, c01 = 0, c02 = 0, c11 = 0, c12 = 0, c22 = 0;
#pragma unroll
    for (int t = 0; t < KNB; ++t) {
        float cx = px[bi[t]] - mx, cy = py[bi[t]] - my, cz = pz[bi[t]] - mz;
        c00 += cx * cx; c01 += cx * cy; c02 += cx * cz;
        c11 += cy * cy; c12 += cy * cz; c22 += cz * cz;
    }
    c00 *= ik; c01 *= ik; c02 *= ik; c11 *= ik; c12 *= ik; c22 *= ik;

    float e[3];
    eig3_sym(c00, c01, c02, c11, c12, c22, e);

    // h = relu(e @ W1 + b1) @ W2 + b2    (W1:[3,4], W2:[4,4] row-major)
    float t4[ED], h4[ED];
#pragma unroll
    for (int jj = 0; jj < ED; ++jj) {
        float s = b1[jj];
#pragma unroll
        for (int ii = 0; ii < 3; ++ii) s += e[ii] * W1[ii * ED + jj];
        t4[jj] = fmaxf(s, 0.0f);
    }
#pragma unroll
    for (int jj = 0; jj < ED; ++jj) {
        float s = b2[jj];
#pragma unroll
        for (int ii = 0; ii < ED; ++ii) s += t4[ii] * W2[ii * ED + jj];
        h4[jj] = s;
    }
#pragma unroll
    for (int jj = 0; jj < ED; ++jj)
        h_out[((size_t)b * NN + q) * ED + jj] = h4[jj];
}

// ------------------- Kernel 2: conv_w -> bf16, K-padded --------------------
__global__ void conv_w_to_bf16(const float* __restrict__ conv_w,
                               unsigned short* __restrict__ wt) {
    int idx = blockIdx.x * blockDim.x + threadIdx.x;    // EMB*CPAD
    if (idx >= EMB * CPAD) return;
    int o = idx / CPAD, c = idx % CPAD;
    wt[idx] = (c < CIN) ? f2bf(conv_w[(size_t)o * CIN + c]) : (unsigned short)0;
}

// ------------------- Kernel 3a: f [B,EMB,N] -> zt [B,N,CPAD] bf16 ----------
// tiled 32x32 transpose through LDS; grid (N/32, EMB/32, B), block (32,8)
__global__ void build_zt_f(const float* __restrict__ f,
                           unsigned short* __restrict__ zt) {
    __shared__ float tile[32][33];
    const int b = blockIdx.z;
    const int c0 = blockIdx.y * 32, n0 = blockIdx.x * 32;
#pragma unroll
    for (int r = 0; r < 32; r += 8)
        tile[threadIdx.y + r][threadIdx.x] =
            f[((size_t)b * EMB + (c0 + threadIdx.y + r)) * NN + n0 + threadIdx.x];
    __syncthreads();
#pragma unroll
    for (int r = 0; r < 32; r += 8) {
        int n = n0 + threadIdx.y + r;
        int c = c0 + threadIdx.x;
        zt[((size_t)b * NN + n) * CPAD + c] = f2bf(tile[threadIdx.x][threadIdx.y + r]);
    }
}

// ------------------- Kernel 3b: h + zero pad into zt tail ------------------
__global__ void build_zt_tail(const float* __restrict__ h,
                              unsigned short* __restrict__ zt) {
    int idx = blockIdx.x * blockDim.x + threadIdx.x;    // B*N*32
    if (idx >= BB * NN * 32) return;
    int j  = idx & 31;
    int pn = idx >> 5;                                  // b*N+n
    float v = (j < ED) ? h[(size_t)pn * ED + j] : 0.0f;
    zt[(size_t)pn * CPAD + EMB + j] = f2bf(v);
}

// ------------------- Kernel 4: WMMA GEMM  y = W @ z + bias -----------------
// grid (N/128, EMB/128, B), block 256 (8 waves).  Wave tile: 32(M) x 64(N).
__global__ void __launch_bounds__(256)
conv_gemm_wmma(const unsigned short* __restrict__ wt,   // [EMB][CPAD] bf16
               const unsigned short* __restrict__ zt,   // [B][N][CPAD] bf16
               const float* __restrict__ conv_b,
               float* __restrict__ y)                   // [B][EMB][N] f32
{
    __shared__ __align__(16) unsigned short wtile[128 * 32];   // [m][k]
    __shared__ __align__(16) unsigned short ztile[128 * 32];   // [n][k]

    const int b    = blockIdx.z;
    const int m0   = blockIdx.y * 128;
    const int n0   = blockIdx.x * 128;
    const int tid  = threadIdx.x;
    const int wave = tid >> 5;
    const int lane = tid & 31;
    const int msub = (wave >> 1) * 32;         // 0/32/64/96 -> M sub-tile base
    const int ngrp = (wave & 1) * 64;          // 0/64       -> N half

    v8f acc[2][4] = {};

    // cooperative staging: thread t owns row = t/2 (0..127), 32B half = t&1
    const int srow = tid >> 1, skc = (tid & 1) * 16;
    const unsigned short* wsrc = wt + (size_t)(m0 + srow) * CPAD + skc;
    const unsigned short* zsrc = zt + ((size_t)b * NN + n0 + srow) * CPAD + skc;

    for (int k0 = 0; k0 < CPAD; k0 += 32) {
        // prefetch next K-step tiles into near caches (global_prefetch_b8)
        if (k0 + 32 < CPAD) {
            __builtin_prefetch(wsrc + k0 + 32, 0, 1);
            __builtin_prefetch(zsrc + k0 + 32, 0, 1);
        }
        // stage W tile 128x32 bf16 (8KB) and Z tile 128x32 bf16 (8KB)
        {
            const float4* s = reinterpret_cast<const float4*>(wsrc + k0);
            float4* d = reinterpret_cast<float4*>(&wtile[srow * 32 + skc]);
            d[0] = s[0]; d[1] = s[1];
        }
        {
            const float4* s = reinterpret_cast<const float4*>(zsrc + k0);
            float4* d = reinterpret_cast<float4*>(&ztile[srow * 32 + skc]);
            d[0] = s[0]; d[1] = s[1];
        }
        __syncthreads();

        // A fragments (contiguous k order, same as B): lanes 0-15 K=0..15,
        // lanes 16-31 K=16..31, row = lane%16
        union { v16bf v; float4 q[2]; } afrag[2];
        const int am = lane & 15, ah = lane >> 4;
#pragma unroll
        for (int mi = 0; mi < 2; ++mi) {
            const int arow = (msub + mi * 16 + am) * 32;
            afrag[mi].q[0] = *reinterpret_cast<const float4*>(&wtile[arow + ah * 16]);
            afrag[mi].q[1] = *reinterpret_cast<const float4*>(&wtile[arow + ah * 16 + 8]);
        }

        // B fragments: col = lane%16, K = (lane/16)*16 + e (contiguous in LDS)
        const int bn = lane & 15, bh = lane >> 4;
#pragma unroll
        for (int j = 0; j < 4; ++j) {
            union { v16bf v; float4 q[2]; } bfrag;
            const float4* bp = reinterpret_cast<const float4*>(
                &ztile[(ngrp + j * 16 + bn) * 32 + bh * 16]);
            bfrag.q[0] = bp[0];
            bfrag.q[1] = bp[1];
            acc[0][j] = __builtin_amdgcn_wmma_f32_16x16x32_bf16(
                false, afrag[0].v, false, bfrag.v, (short)0, acc[0][j], false, false);
            acc[1][j] = __builtin_amdgcn_wmma_f32_16x16x32_bf16(
                false, afrag[1].v, false, bfrag.v, (short)0, acc[1][j], false, false);
        }
        __syncthreads();
    }

    // epilogue: C/D layout -> lane 0-15: M=r, lane 16-31: M=r+8, N=lane%16
    const int cl = lane & 15, ch = lane >> 4;
#pragma unroll
    for (int mi = 0; mi < 2; ++mi) {
#pragma unroll
        for (int j = 0; j < 4; ++j) {
#pragma unroll
            for (int r = 0; r < 8; ++r) {
                int m = m0 + msub + mi * 16 + r + ch * 8;
                int n = n0 + ngrp + j * 16 + cl;
                y[((size_t)b * EMB + m) * NN + n] = acc[mi][j][r] + conv_b[m];
            }
        }
    }
}

// ------------------- Kernel 5: BN statistics per channel -------------------
__global__ void __launch_bounds__(256)
bn_stats(const float* __restrict__ y, float* __restrict__ stats) {
    __shared__ float ssum[256], ssq[256];
    const int o = blockIdx.x;
    float s = 0.f, q = 0.f;
    for (int i = threadIdx.x; i < BB * NN; i += 256) {
        int b = i >> 12, n = i & (NN - 1);
        float v = y[((size_t)b * EMB + o) * NN + n];
        s += v; q += v * v;
    }
    ssum[threadIdx.x] = s; ssq[threadIdx.x] = q;
    __syncthreads();
    for (int st = 128; st > 0; st >>= 1) {
        if (threadIdx.x < st) {
            ssum[threadIdx.x] += ssum[threadIdx.x + st];
            ssq[threadIdx.x]  += ssq[threadIdx.x + st];
        }
        __syncthreads();
    }
    if (threadIdx.x == 0) {
        float mean = ssum[0] * (1.0f / (BB * NN));
        float var  = ssq[0] * (1.0f / (BB * NN)) - mean * mean;
        stats[o * 2 + 0] = mean;
        stats[o * 2 + 1] = rsqrtf(var + BN_EPS);
    }
}

// ------------------- Kernel 6: BN apply + ReLU (in place) ------------------
__global__ void __launch_bounds__(256)
bn_apply_relu(float* __restrict__ y, const float* __restrict__ stats,
              const float* __restrict__ gamma, const float* __restrict__ beta) {
    size_t idx = (size_t)blockIdx.x * blockDim.x + threadIdx.x;  // B*EMB*N
    if (idx >= (size_t)BB * EMB * NN) return;
    int o = (int)((idx >> 12) & (EMB - 1));
    float v = y[idx];
    v = (v - stats[o * 2]) * stats[o * 2 + 1] * gamma[o] + beta[o];
    y[idx] = fmaxf(v, 0.0f);
}

// ---------------------------------------------------------------------------
extern "C" void kernel_launch(void* const* d_in, const int* in_sizes, int n_in,
                              void* d_out, int out_size, void* d_ws, size_t ws_size,
                              hipStream_t stream) {
    const float* data   = (const float*)d_in[0];
    const float* f      = (const float*)d_in[1];
    const float* W1     = (const float*)d_in[2];
    const float* b1     = (const float*)d_in[3];
    const float* W2     = (const float*)d_in[4];
    const float* b2     = (const float*)d_in[5];
    const float* conv_w = (const float*)d_in[6];
    const float* conv_b = (const float*)d_in[7];
    const float* gamma  = (const float*)d_in[8];
    const float* beta   = (const float*)d_in[9];
    (void)in_sizes; (void)n_in; (void)out_size; (void)ws_size;

    float* out      = (float*)d_out;
    float* data_out = out;                        // B*N*3
    float* y        = out + (size_t)BB * NN * 3;  // B*EMB*N (pre-BN, then final)

    // workspace layout (bytes): h | wt | zt | stats  (~37 MB total)
    char* ws = (char*)d_ws;
    float*          h     = (float*)ws;                                   // B*N*4 f32
    unsigned short* wt    = (unsigned short*)(ws + 262144);               // EMB*CPAD bf16
    unsigned short* zt    = (unsigned short*)(ws + 262144 + 2162688);     // B*N*CPAD bf16
    float*          stats = (float*)(ws + 262144 + 2162688 + 34603008);   // EMB*2 f32

    // 1) fused KNN / covariance / eigvals / MLP (+data passthrough)
    knn_eig_mlp<<<dim3(NN / 256, BB), 256, 0, stream>>>(
        data, W1, b1, W2, b2, h, data_out);

    // 2) weight conversion to bf16 (K-padded)
    conv_w_to_bf16<<<(EMB * CPAD + 255) / 256, 256, 0, stream>>>(conv_w, wt);

    // 3) point-major bf16 z: transpose f, then h tail + zero pad
    build_zt_f<<<dim3(NN / 32, EMB / 32, BB), dim3(32, 8), 0, stream>>>(f, zt);
    build_zt_tail<<<(BB * NN * 32 + 255) / 256, 256, 0, stream>>>(h, zt);

    // 4) WMMA GEMM  y = W @ z + bias
    conv_gemm_wmma<<<dim3(NN / 128, EMB / 128, BB), 256, 0, stream>>>(
        wt, zt, conv_b, y);

    // 5/6) BatchNorm (training stats) + ReLU
    bn_stats<<<EMB, 256, 0, stream>>>(y, stats);
    bn_apply_relu<<<(int)(((size_t)BB * EMB * NN + 255) / 256), 256, 0, stream>>>(
        y, stats, gamma, beta);
}
